// NonLearnableLayer_67989332296197
// MI455X (gfx1250) — compile-verified, matrix operand order checked by reading, not run on hardware
//
#include <hip/hip_runtime.h>

typedef unsigned int   u32;
typedef unsigned short u16;
typedef __attribute__((ext_vector_type(16))) __bf16 v16bf;
typedef __attribute__((ext_vector_type(2)))  __bf16 v2bf;
typedef __attribute__((ext_vector_type(8)))  float  v8f;
typedef __attribute__((ext_vector_type(4)))  u32    u32x4;

union Frag {          // one WMMA 16x32 bf16 operand fragment (per-lane view)
  v16bf v;
  u32x4 q[2];
  u32   u[8];
};
union Pk {            // 8 bf16 packed for a 16B store
  u32x4 q;
  u32   u[4];
};

__device__ __forceinline__ u32 pack2bf(float a, float b) {  // -> v_cvt_pk_bf16_f32
  union { v2bf v; u32 u; } p;
  p.v[0] = (__bf16)a; p.v[1] = (__bf16)b;
  return p.u;
}
__device__ __forceinline__ u16 f2bf(float f) {
  union { __bf16 h; u16 u; } p; p.h = (__bf16)f; return p.u;
}
__device__ __forceinline__ float bf2f(u16 s) {
  return __uint_as_float(((u32)s) << 16);
}

constexpr int CIN = 192, COUT = 192, HH = 56, WW = 56;
constexpr int KT1 = 54;                 // 1728 / 32 K-tiles (conv GEMM)
constexpr int KT2 = 18;                 // 576 / 32 K-tiles (1x1 GEMM)
constexpr int MT  = 12;                 // 192 / 16 M-tiles
constexpr int BLK = 256;                // 8 waves: (4 M-groups) x (2 N-pairs)
constexpr int XS_U32   = (3 * 58) * (CIN / 2);        // x stage: 174 cols * 96 dwords
constexpr int XCAT_STRIDE_H = 592;                    // 576 + 16 pad halves per column
constexpr int XCAT_U32 = 64 * (XCAT_STRIDE_H / 2);
constexpr int SMEM_U32 = XS_U32 + XCAT_U32;           // 35648 dwords = 142592 B
constexpr int A2_OFF_H = MT * KT1 * 32 * 16;          // halves offset of w_lin frags

// ---- weight pre-swizzle: pack fp32 weights into exact WMMA A-fragment layout ----
// fragment (mt,kt): 32 lanes x 16 bf16 (32B contiguous per lane).
// lane l (l%16 = row M within tile), element e -> K_local = e + (e&8) + (l>=16 ? 8 : 0)

__global__ void prep_wconv(const float* __restrict__ w, u16* __restrict__ dst) {
  int t = blockIdx.x * blockDim.x + threadIdx.x;
  if (t >= MT * KT1 * 32) return;
  int lane = t & 31, frag = t >> 5;
  int kt = frag % KT1, mt = frag / KT1;
  int m  = mt * 16 + (lane & 15);
  int hi = lane >> 4;
  float v[16];
  #pragma unroll
  for (int e = 0; e < 16; ++e) {
    int klocal = e + (e & 8) + hi * 8;
    int k  = kt * 32 + klocal;          // K = patch*192 + ci
    int ci = k % CIN;
    int patch = k / CIN;                // 0..8 = kh*3+kw
    v[e] = w[(m * CIN + ci) * 9 + patch];
  }
  Frag f;
  #pragma unroll
  for (int j = 0; j < 8; ++j) f.u[j] = pack2bf(v[2 * j], v[2 * j + 1]);
  u32x4* dp = (u32x4*)(dst + (size_t)(frag * 32 + lane) * 16);
  dp[0] = f.q[0]; dp[1] = f.q[1];
}

__global__ void prep_wlin(const float* __restrict__ w, u16* __restrict__ dst) {
  int t = blockIdx.x * blockDim.x + threadIdx.x;
  if (t >= MT * KT2 * 32) return;
  int lane = t & 31, frag = t >> 5;
  int kt = frag % KT2, mt = frag / KT2;
  int m  = mt * 16 + (lane & 15);
  int hi = lane >> 4;
  float v[16];
  #pragma unroll
  for (int e = 0; e < 16; ++e) {
    int klocal = e + (e & 8) + hi * 8;
    int k = kt * 32 + klocal;           // 0..575
    v[e] = w[m * 576 + k];
  }
  Frag f;
  #pragma unroll
  for (int j = 0; j < 8; ++j) f.u[j] = pack2bf(v[2 * j], v[2 * j + 1]);
  u32x4* dp = (u32x4*)(dst + (size_t)(frag * 32 + lane) * 16);
  dp[0] = f.q[0]; dp[1] = f.q[1];
}

// ---- fused: conv3x3(bin) + relu + maxpool + avgpool + 1x1 GEMM, one (b, h) row per WG ----
__global__ __launch_bounds__(BLK) void lbconv_fused(const float* __restrict__ x,
                                                    const u16*  __restrict__ afrag,
                                                    float* __restrict__ out) {
  extern __shared__ u32 smem[];
  u16* xs16   = (u16*)smem;             // staged x, layout [(r*58+col)][192] bf16
  u32* xcatU  = smem + XS_U32;          // concat buffer, layout [n][576+pad] bf16
  u16* xcat16 = (u16*)xcatU;

  const int tid = threadIdx.x;
  const int b = blockIdx.x / HH;
  const int h = blockIdx.x % HH;

  // zero the stage region: gives conv/avg-pool zero padding + missing boundary rows
  for (int i = tid; i < XS_U32; i += BLK) smem[i] = 0;
  __syncthreads();

  // stage x[b, :, h-1..h+1, :] -> bf16, channel-innermost (coalesced global reads)
  for (int i = tid; i < CIN * 3 * WW; i += BLK) {
    int col = i % WW;
    int r   = (i / WW) % 3;
    int ci  = i / (WW * 3);
    int gr  = h - 1 + r;
    if (gr >= 0 && gr < HH) {
      float v = x[(((size_t)b * CIN + ci) * HH + gr) * WW + col];
      xs16[(r * 58 + col + 1) * CIN + ci] = f2bf(v);
    }
  }
  __syncthreads();

  const int lane = tid & 31;
  const int wave = tid >> 5;
  const int mg   = wave >> 1;           // M group: 3 M-tiles each
  const int nh   = wave & 1;            // N pair: 2 N-tiles each
  const int hi   = lane >> 4;
  const int lmod = lane & 15;

  int narr[2], nnarr[2];
  #pragma unroll
  for (int j = 0; j < 2; ++j) {
    narr[j]  = (2 * nh + j) * 16 + lmod;
    nnarr[j] = (narr[j] < WW) ? narr[j] : (WW - 1);  // clamp pad lanes (discarded)
  }

  v8f acc[3][2] = {};

  // GEMM1: x1 = conv(x, w_conv), implicit GEMM M=192 K=1728 N=64.
  // Each wave: 3 M-tiles x 2 N-tiles => 6 wmma per K-step, A loaded once per M-tile.
  for (int kt = 0; kt < KT1; ++kt) {
    int patch = kt / 6;                 // 192%32==0 -> one 3x3 tap per K-tile
    int kw  = patch % 3;
    int row = patch / 3;
    int ci0 = (kt % 6) * 32;
    Frag bfr[2];
    #pragma unroll
    for (int j = 0; j < 2; ++j) {
      int baseu = (row * 58 + nnarr[j] + kw) * (CIN / 2) + (ci0 >> 1) + hi * 4;
      __builtin_assume((baseu & 3) == 0);    // 16B-aligned -> ds_load_b128
      bfr[j].q[0] = *(const u32x4*)(smem + baseu);
      bfr[j].q[1] = *(const u32x4*)(smem + baseu + 8);
    }
    __builtin_prefetch(afrag + (size_t)((mg * 3 * KT1 + kt + 1) * 32 + lane) * 16, 0, 3);
    #pragma unroll
    for (int t3 = 0; t3 < 3; ++t3) {
      int mt = mg * 3 + t3;
      const u32x4* ap = (const u32x4*)(afrag + (size_t)((mt * KT1 + kt) * 32 + lane) * 16);
      Frag afr; afr.q[0] = ap[0]; afr.q[1] = ap[1];
      #pragma unroll
      for (int j = 0; j < 2; ++j) {
        acc[t3][j] = __builtin_amdgcn_wmma_f32_16x16x32_bf16(
            false, afr.v, false, bfr[j].v, (short)0, acc[t3][j], false, false);
      }
    }
  }

  // ReLU + bf16 + scatter x1 into xcat rows 0..191 (D layout: lane=N, vgpr e -> M)
  #pragma unroll
  for (int t3 = 0; t3 < 3; ++t3) {
    int m0 = (mg * 3 + t3) * 16 + hi * 8;
    #pragma unroll
    for (int j = 0; j < 2; ++j) {
      Pk pk;
      #pragma unroll
      for (int q = 0; q < 4; ++q) {
        pk.u[q] = pack2bf(fmaxf(acc[t3][j][2 * q],     0.0f),
                          fmaxf(acc[t3][j][2 * q + 1], 0.0f));
      }
      *(u32x4*)(xcatU + narr[j] * (XCAT_STRIDE_H / 2) + (m0 >> 1)) = pk.q;
    }
  }

  // pools: max (-inf padding semantics via validity) + avg (count_include_pad via zeros)
  {
    const bool r0ok = (h > 0), r2ok = (h < HH - 1);
    for (int i = tid; i < CIN * WW; i += BLK) {
      int n2 = i % WW;
      int c  = i / WW;
      float mx = -3.402823466e38f;
      float s  = 0.0f;
      #pragma unroll
      for (int dr = 0; dr < 3; ++dr) {
        bool rok = (dr == 1) || (dr == 0 ? r0ok : r2ok);
        #pragma unroll
        for (int dc = 0; dc < 3; ++dc) {
          float v = bf2f(xs16[(dr * 58 + n2 + dc) * CIN + c]);
          s += v;
          int cc = n2 + dc - 1;
          if (rok && cc >= 0 && cc < WW) mx = fmaxf(mx, v);
        }
      }
      xcat16[n2 * XCAT_STRIDE_H + CIN + c]     = f2bf(mx);
      xcat16[n2 * XCAT_STRIDE_H + 2 * CIN + c] = f2bf(s * (1.0f / 9.0f));
    }
  }
  __syncthreads();

  // GEMM2: out = w_lin[192x576] @ xcat[576x64]
  #pragma unroll
  for (int t3 = 0; t3 < 3; ++t3)
    #pragma unroll
    for (int j = 0; j < 2; ++j) acc[t3][j] = (v8f){};
  const u16* a2 = afrag + A2_OFF_H;
  for (int kt = 0; kt < KT2; ++kt) {
    Frag bfr[2];
    #pragma unroll
    for (int j = 0; j < 2; ++j) {
      int baseu = narr[j] * (XCAT_STRIDE_H / 2) + kt * 16 + hi * 4;
      __builtin_assume((baseu & 3) == 0);
      bfr[j].q[0] = *(const u32x4*)(xcatU + baseu);
      bfr[j].q[1] = *(const u32x4*)(xcatU + baseu + 8);
    }
    #pragma unroll
    for (int t3 = 0; t3 < 3; ++t3) {
      int mt = mg * 3 + t3;
      const u32x4* ap = (const u32x4*)(a2 + (size_t)((mt * KT2 + kt) * 32 + lane) * 16);
      Frag afr; afr.q[0] = ap[0]; afr.q[1] = ap[1];
      #pragma unroll
      for (int j = 0; j < 2; ++j) {
        acc[t3][j] = __builtin_amdgcn_wmma_f32_16x16x32_bf16(
            false, afr.v, false, bfr[j].v, (short)0, acc[t3][j], false, false);
      }
    }
  }

  #pragma unroll
  for (int j = 0; j < 2; ++j) {
    if (narr[j] < WW) {
      #pragma unroll
      for (int t3 = 0; t3 < 3; ++t3) {
        int m0 = (mg * 3 + t3) * 16 + hi * 8;
        #pragma unroll
        for (int e = 0; e < 8; ++e) {
          out[(((size_t)b * COUT + (m0 + e)) * HH + h) * WW + narr[j]] = acc[t3][j][e];
        }
      }
    }
  }
}

extern "C" void kernel_launch(void* const* d_in, const int* in_sizes, int n_in,
                              void* d_out, int out_size, void* d_ws, size_t ws_size,
                              hipStream_t stream) {
  const float* x      = (const float*)d_in[0];
  const float* w_conv = (const float*)d_in[1];
  const float* w_lin  = (const float*)d_in[2];
  float* out = (float*)d_out;
  u16* afrag = (u16*)d_ws;              // w_conv frags (663KB) then w_lin frags (221KB)

  int t1 = MT * KT1 * 32;
  prep_wconv<<<(t1 + 127) / 128, 128, 0, stream>>>(w_conv, afrag);
  int t2 = MT * KT2 * 32;
  prep_wlin<<<(t2 + 127) / 128, 128, 0, stream>>>(w_lin, afrag + A2_OFF_H);

  lbconv_fused<<<dim3(32 * HH), dim3(BLK), SMEM_U32 * sizeof(u32), stream>>>(x, afrag, out);
}
